// LinearCRF_5334349382141
// MI455X (gfx1250) — compile-verified, matrix operand order checked by reading, not run on hardware
//
#include <hip/hip_runtime.h>
#include <hip/hip_bf16.h>

typedef __attribute__((ext_vector_type(16))) _Float16 v16h;
typedef __attribute__((ext_vector_type(8)))  _Float16 v8h;
typedef __attribute__((ext_vector_type(8)))  float    v8f;

#define B_TOT   32
#define T_LEN   256
#define L_LAB   128
#define START_  125
#define END_    126
#define ASTRIDE 129   // f32 alpha row stride (pad for banks)
#define VSTRIDE 136   // f16 exp buffer row stride (keeps 16B alignment: 272B = 17*16)

// One block = 16 batches (one WMMA M-tile). 8 waves; wave w owns N columns [16w,16w+16).
__global__ __launch_bounds__(256) void crf_fwd_kernel(
    const float* __restrict__ lstm,    // [32,256,128]
    const float* __restrict__ trans,   // [128,128]
    const int*   __restrict__ lens,    // [32]
    const int*   __restrict__ tags,    // [32,256]
    float* __restrict__ out)
{
    __shared__ __align__(16) float     alpha[16 * ASTRIDE];
    __shared__ __align__(16) _Float16  vbuf [16 * VSTRIDE];
    __shared__ float mbuf[16];
    __shared__ float ubuf[16];
    __shared__ float lbuf[16];

    const int tid   = threadIdx.x;
    const int lane  = tid & 31;
    const int wave  = tid >> 5;
    const int bbase = blockIdx.x * 16;
    const int khalf = ((lane >> 4) & 1) << 3;   // lanes 16-31 hold K+8 per the 16-bit A/B layout

    // ---- B fragments: E = exp(trans), constant for the whole scan, resident in VGPRs.
    // Wave w covers columns n = 16w + (lane&15); element e -> K = c*32 + (e&7) + (e>>3)*16 + khalf.
    const int ncol = (wave << 4) + (lane & 15);
    v16h Bf[4];
#pragma unroll
    for (int c = 0; c < 4; ++c) {
#pragma unroll
        for (int e = 0; e < 16; ++e) {
            int k = (c << 5) + (e & 7) + ((e >> 3) << 4) + khalf;
            Bf[c][e] = (_Float16)__expf(trans[k * L_LAB + ncol]);
        }
    }

    // Sequence lengths for the rows this lane writes in the D phase.
    int lenreg[8];
#pragma unroll
    for (int r = 0; r < 8; ++r) {
        int M = r + (((lane >> 4) & 1) << 3);
        lenreg[r] = lens[bbase + M];
    }

    // ---- alpha0[b][j] = trans[START][j] + lstm[b][0][j]
    for (int idx = tid; idx < 16 * L_LAB; idx += 256) {
        int b = idx >> 7, j = idx & 127;
        alpha[b * ASTRIDE + j] =
            trans[START_ * L_LAB + j] + lstm[(size_t)(bbase + b) * T_LEN * L_LAB + j];
    }
    __syncthreads();

    // ---- sequential scan over T (latency-critical path)
    for (int t = 1; t < T_LEN; ++t) {
        // Phase 1: per-row max + exp into f16 LDS buffer. Wave w handles rows 2w, 2w+1.
#pragma unroll
        for (int rr = 0; rr < 2; ++rr) {
            int r = (wave << 1) + rr;
            float a0 = alpha[r * ASTRIDE + lane];
            float a1 = alpha[r * ASTRIDE + lane + 32];
            float a2 = alpha[r * ASTRIDE + lane + 64];
            float a3 = alpha[r * ASTRIDE + lane + 96];
            float m = fmaxf(fmaxf(a0, a1), fmaxf(a2, a3));
#pragma unroll
            for (int off = 16; off > 0; off >>= 1)
                m = fmaxf(m, __shfl_xor(m, off, 32));
            vbuf[r * VSTRIDE + lane]      = (_Float16)__expf(a0 - m);
            vbuf[r * VSTRIDE + lane + 32] = (_Float16)__expf(a1 - m);
            vbuf[r * VSTRIDE + lane + 64] = (_Float16)__expf(a2 - m);
            vbuf[r * VSTRIDE + lane + 96] = (_Float16)__expf(a3 - m);
            if (lane == 0) mbuf[r] = m;
        }
        __syncthreads();

        // Phase 2: W[0:16, ntile] = V(16x128,f16) @ E(128x16,f16), f32 accumulate.
        v8f acc = {};
        const int arow = lane & 15;
#pragma unroll
        for (int c = 0; c < 4; ++c) {
            // A fragment: two aligned 16B LDS reads per the 16-bit 16x32 layout.
            const v8h* pa = (const v8h*)&vbuf[arow * VSTRIDE + (c << 5) + khalf];
            v8h lo = pa[0];
            v8h hi = pa[2];
            v16h a;
#pragma unroll
            for (int e = 0; e < 8; ++e) { a[e] = lo[e]; a[e + 8] = hi[e]; }
            acc = __builtin_amdgcn_wmma_f32_16x16x32_f16(
                false, a, false, Bf[c], (short)0, acc, false, false);
        }

        // Phase 3: alpha_t = m + log(W) + lstm[:,t,:], written straight from D fragments.
        const int j = (wave << 4) + (lane & 15);
#pragma unroll
        for (int r = 0; r < 8; ++r) {
            int M = r + (((lane >> 4) & 1) << 3);
            float w  = acc[r];
            float an = mbuf[M] + __logf(fmaxf(w, 1e-30f))
                     + lstm[((size_t)(bbase + M) * T_LEN + t) * L_LAB + j];
            if (t < lenreg[r]) alpha[M * ASTRIDE + j] = an;    // freeze at alpha[len-1]
            if (t + 1 < T_LEN)
                __builtin_prefetch(&lstm[((size_t)(bbase + M) * T_LEN + t + 1) * L_LAB + j], 0, 1);
        }
        __syncthreads();
    }

    // ---- unlabeled[b] = logsumexp_j( alpha[len-1][b,j] + trans[j,END] )
#pragma unroll
    for (int rr = 0; rr < 2; ++rr) {
        int r = (wave << 1) + rr;
        float v0 = alpha[r * ASTRIDE + lane]      + trans[(lane)      * L_LAB + END_];
        float v1 = alpha[r * ASTRIDE + lane + 32] + trans[(lane + 32) * L_LAB + END_];
        float v2 = alpha[r * ASTRIDE + lane + 64] + trans[(lane + 64) * L_LAB + END_];
        float v3 = alpha[r * ASTRIDE + lane + 96] + trans[(lane + 96) * L_LAB + END_];
        float m = fmaxf(fmaxf(v0, v1), fmaxf(v2, v3));
#pragma unroll
        for (int off = 16; off > 0; off >>= 1)
            m = fmaxf(m, __shfl_xor(m, off, 32));
        float s = __expf(v0 - m) + __expf(v1 - m) + __expf(v2 - m) + __expf(v3 - m);
#pragma unroll
        for (int off = 16; off > 0; off >>= 1)
            s += __shfl_xor(s, off, 32);
        if (lane == 0) ubuf[r] = m + __logf(s);
    }

    // ---- labeled path score (gathers; one lane per batch row)
    if (tid < 16) {
        int b   = bbase + tid;
        int len = lens[b];
        int tg0 = tags[b * T_LEN];
        float sc = lstm[(size_t)b * T_LEN * L_LAB + tg0] + trans[START_ * L_LAB + tg0];
        int prev = tg0;
        for (int t = 1; t < len; ++t) {
            int cur = tags[b * T_LEN + t];
            sc += trans[prev * L_LAB + cur] + lstm[((size_t)b * T_LEN + t) * L_LAB + cur];
            prev = cur;
        }
        sc += trans[prev * L_LAB + END_];
        lbuf[tid] = sc;
    }
    __syncthreads();

    if (tid == 0) {
        float p = 0.f;
#pragma unroll
        for (int r = 0; r < 16; ++r) p += ubuf[r] - lbuf[r];
        atomicAdd(out, p * (1.0f / (float)B_TOT));
    }
}

extern "C" void kernel_launch(void* const* d_in, const int* in_sizes, int n_in,
                              void* d_out, int out_size, void* d_ws, size_t ws_size,
                              hipStream_t stream) {
    const float* lstm  = (const float*)d_in[0];  // lstm_scores [32,256,128] f32
    const float* trans = (const float*)d_in[1];  // transition  [128,128]    f32
    const int*   lens  = (const int*)d_in[2];    // word_seq_lens [32]       i32
    const int*   tags  = (const int*)d_in[3];    // tags [32,256]            i32
    // d_in[4] = mask: recomputed from lens, unused.
    float* out = (float*)d_out;

    hipMemsetAsync(out, 0, sizeof(float), stream);   // capture-safe
    hipLaunchKernelGGL(crf_fwd_kernel, dim3(2), dim3(256), 0, stream,
                       lstm, trans, lens, tags, out);
}
